// GNNCrossAttentionHead_6923487281729
// MI455X (gfx1250) — matrix operand order repaired and checked by reading, not compiled, not run on hardware
//
#include <hip/hip_runtime.h>
#include <hip/hip_bf16.h>
#include <cstdint>
#include <cstddef>

// ---------------- problem constants ----------------
#define BB     16
#define GNNF   256
#define HD     512
#define CTXK   4      // context length (K in reference)
#define NHD    8
#define DHD    64
#define NGENE  6640
#define NCLS   3
#define INNERD 2048
#define MT     64                         // gene rows per block tile
#define TILES  ((NGENE + MT - 1) / MT)    // 104 (last tile partial)

typedef __attribute__((ext_vector_type(16))) __bf16 v16bf;
typedef __attribute__((ext_vector_type(8)))  float  v8f;
typedef __attribute__((ext_vector_type(4)))  unsigned int v4u;
typedef __attribute__((ext_vector_type(4)))  int v4i;
typedef __attribute__((ext_vector_type(8)))  int v8i;

#if __has_builtin(__builtin_amdgcn_tensor_load_to_lds) && __has_builtin(__builtin_amdgcn_s_wait_tensorcnt)
#define HAVE_TDM 1
#else
#define HAVE_TDM 0
#endif

// ---------------- small helpers ----------------
__device__ __forceinline__ float gelu_f(float x) {
  return 0.5f * x * (1.0f + erff(x * 0.70710678118654752f));
}

// 256-thread block sum (8 waves)
__device__ __forceinline__ float block_sum256(float v, float* red8) {
#pragma unroll
  for (int m = 1; m < 32; m <<= 1) v += __shfl_xor(v, m);
  int wv = threadIdx.x >> 5, ln = threadIdx.x & 31;
  if (ln == 0) red8[wv] = v;
  __syncthreads();
  float s = 0.f;
#pragma unroll
  for (int i = 0; i < 8; ++i) s += red8[i];
  __syncthreads();
  return s;
}

// A-fragment (16x32 bf16) element placement, per CDNA5 ISA 7.12.2:
// lanes 0-15 hold K=0..7 (slots 0-7) and K=16..23 (slots 8-15); lanes 16-31 hold K=8..15, K=24..31.
__device__ __forceinline__ int afidx(int mt, int kt, int m, int k) {
  int lane = m + (((k & 15) >= 8) ? 16 : 0);
  int slot = (k & 7) + ((k >> 4) << 3);
  return (mt * 16 + kt) * 512 + lane * 16 + slot;
}

// ---------------- TDM: stage K/V block (CTXK*2*HD f32 = 16KB) into LDS ----------------
// D# per CDNA5 ISA 08_async_tensor §8: group0 = {count/flags, lds_addr, global_addr, type},
// group1 = {mask/data_size, dims/strides}, groups 2-3 zero (<=2D tensor).
// 6-arg builtin form (clang-23 / therock-10.0 toolchain).
__device__ __forceinline__ void tdm_issue_kv(float* ldsKV, const float* __restrict__ kvb) {
#if HAVE_TDM
  if (threadIdx.x < 32) {
    const unsigned NE = CTXK * 2 * HD;  // 4096 elements (f32)
    unsigned long long ga = (unsigned long long)(uintptr_t)kvb;  // 57-bit global byte addr
    unsigned la = (unsigned)(uintptr_t)ldsKV;                    // LDS byte addr = addr[31:0]
    v4u g0 = { 1u,                                  // count=1 (valid user descriptor)
               la,                                  // lds_addr [63:32]
               (unsigned)ga,                        // global_addr [95:64]
               (unsigned)(ga >> 32) | (2u << 30) }; // global_addr[56:32] | type=2 ("image")
    v8i g1 = { (int)0x00020000u,                    // workgroup_mask=0, data_size=2 (4 bytes)
               (int)((NE & 0xFFFFu) << 16),         // tensor_dim0[15:0] -> bits[63:48]
               (int)((NE >> 16) | (1u << 16)),      // tensor_dim0[31:16], tensor_dim1=1
               (int)((NE & 0xFFFFu) << 16),         // tile_dim0 -> bits[127:112]
               1,                                   // tile_dim1=1 (tile_dim2=0)
               (int)NE,                             // tensor_dim0_stride[31:0]
               0, 0 };                              // stride hi / tensor_dim1_stride = 0
    v4i g2 = { 0, 0, 0, 0 };
    v4i g3 = { 0, 0, 0, 0 };
    v8i g4 = { 0, 0, 0, 0, 0, 0, 0, 0 };
    __builtin_amdgcn_tensor_load_to_lds(g0, g1, g2, g3, g4, 0);
  }
#else
  for (int i = threadIdx.x; i < CTXK * 2 * HD; i += 256) ldsKV[i] = kvb[i];
#endif
}

__device__ __forceinline__ void tdm_wait_kv() {
#if HAVE_TDM
  if (threadIdx.x < 32) __builtin_amdgcn_s_wait_tensorcnt(0);
#endif
}

// ---------------- weight pre-pack: f32 [512,512] row-major -> bf16 B-fragment tiles ----------------
// B fragment (32x16): lane = n + 16*((k/8)%2), slot = k%8 + 8*(k/16); fragment (nt,kt) is 512 contiguous bf16.
__global__ __launch_bounds__(256) void k_pack_w(const float* __restrict__ src,
                                                __bf16* __restrict__ dst) {
  int idx = blockIdx.x * 256 + threadIdx.x;   // idx = n*512 + k
  int n = idx >> 9, k = idx & 511;
  int nt = n >> 4, nn = n & 15;
  int kt = k >> 5, kk = k & 31;
  int lane = nn + (((kk >> 3) & 1) << 4);
  int slot = (kk & 7) + ((kk >> 4) << 3);
  dst[(nt * 16 + kt) * 512 + lane * 16 + slot] = (__bf16)src[idx];
}

// ---------------- LN rows -> LDS A-fragment bf16 tile (64 rows x 512 cols) ----------------
__device__ __forceinline__ void ln_rows_to_ldsA(__bf16* ldsA, const float* __restrict__ base,
                                                int row0, int nrows,
                                                const float* __restrict__ g,
                                                const float* __restrict__ bt) {
  int wv = threadIdx.x >> 5, ln = threadIdx.x & 31;
  int c0 = ln * 16;
#pragma unroll
  for (int i = 0; i < 8; ++i) {
    int rl = wv * 8 + i;
    int rg = row0 + rl;
    int rc = rg < nrows ? rg : nrows - 1;
    const float* src = base + (size_t)rc * HD;
    float vals[16];
    float s = 0.f, ss = 0.f;
#pragma unroll
    for (int j = 0; j < 16; ++j) { float x = src[c0 + j]; vals[j] = x; s += x; ss += x * x; }
#pragma unroll
    for (int m = 1; m < 32; m <<= 1) { s += __shfl_xor(s, m); ss += __shfl_xor(ss, m); }
    float mean = s * (1.f / HD);
    float inv  = rsqrtf(ss * (1.f / HD) - mean * mean + 1e-5f);
    int mt = rl >> 4, mm = rl & 15;
#pragma unroll
    for (int j = 0; j < 16; ++j) {
      int kg = c0 + j;
      float y = (vals[j] - mean) * inv * g[kg] + bt[kg];
      ldsA[afidx(mt, kg >> 5, mm, kg & 31)] = (__bf16)y;
    }
  }
}

// ---------------- core WMMA GEMM: [64 x 512] (LDS, bf16) x [512 x 512] (packed B, bf16) ----------------
// wave w owns output columns [64w, 64w+64): 4 M-tiles x 4 N-tiles, f32 accum.
__device__ __forceinline__ void gemm512(const __bf16* ldsA, const __bf16* __restrict__ wp,
                                        v8f acc[4][4]) {
  int wv = threadIdx.x >> 5, ln = threadIdx.x & 31;
  for (int kt = 0; kt < 16; ++kt) {
    v16bf bfr[4];
#pragma unroll
    for (int n2 = 0; n2 < 4; ++n2)
      bfr[n2] = *(const v16bf*)(wp + ((wv * 4 + n2) * 16 + kt) * 512 + ln * 16);
    if (kt + 1 < 16)
      __builtin_prefetch(wp + ((wv * 4) * 16 + kt + 1) * 512 + ln * 16, 0, 1);
#pragma unroll
    for (int mt = 0; mt < 4; ++mt) {
      v16bf afr = *(const v16bf*)(ldsA + (mt * 16 + kt) * 512 + ln * 16);
#pragma unroll
      for (int n2 = 0; n2 < 4; ++n2)
        acc[mt][n2] = __builtin_amdgcn_wmma_f32_16x16x32_bf16(
            false, afr, false, bfr[n2], (short)0, acc[mt][n2], false, false);
    }
  }
}

// ---------------- attention over 4 keys, wave == one head (cols 64w..64w+63) ----------------
// acc holds Q in C layout (M = j + 8*(lane/16), N = lane%16 per 16x16 tile); overwritten with o.
// kvb points at the LDS-staged K/V block: [kk][sel][d].
__device__ __forceinline__ void attn4(v8f acc[4][4], const float* kvb) {
  int wv = threadIdx.x >> 5, ln = threadIdx.x & 31;
  int nl = ln & 15;
  float kvl[4][4], vvl[4][4];
#pragma unroll
  for (int kk = 0; kk < 4; ++kk)
#pragma unroll
    for (int n2 = 0; n2 < 4; ++n2) {
      int d = wv * 64 + n2 * 16 + nl;
      kvl[kk][n2] = kvb[(kk * 2 + 0) * HD + d];
      vvl[kk][n2] = kvb[(kk * 2 + 1) * HD + d];
    }
#pragma unroll
  for (int mt = 0; mt < 4; ++mt)
#pragma unroll
    for (int j = 0; j < 8; ++j) {
      float sc[4];
#pragma unroll
      for (int kk = 0; kk < 4; ++kk) {
        float p = acc[mt][0][j] * kvl[kk][0] + acc[mt][1][j] * kvl[kk][1]
                + acc[mt][2][j] * kvl[kk][2] + acc[mt][3][j] * kvl[kk][3];
        // reduce across the 16-lane half (covers full DH=64 with the 4 n-tiles)
        p += __shfl_xor(p, 1); p += __shfl_xor(p, 2);
        p += __shfl_xor(p, 4); p += __shfl_xor(p, 8);
        sc[kk] = p * 0.125f;  // 1/sqrt(64)
      }
      float mx = fmaxf(fmaxf(sc[0], sc[1]), fmaxf(sc[2], sc[3]));
      float e0 = __expf(sc[0] - mx), e1 = __expf(sc[1] - mx);
      float e2 = __expf(sc[2] - mx), e3 = __expf(sc[3] - mx);
      float inv = 1.f / (e0 + e1 + e2 + e3);
#pragma unroll
      for (int n2 = 0; n2 < 4; ++n2)
        acc[mt][n2][j] = (e0 * vvl[0][n2] + e1 * vvl[1][n2]
                        + e2 * vvl[2][n2] + e3 * vvl[3][n2]) * inv;
    }
}

// store o (C layout f32) into LDS as bf16 A fragments for the O-projection GEMM
__device__ __forceinline__ void store_o_ldsA(__bf16* ldsA, v8f acc[4][4]) {
  int wv = threadIdx.x >> 5, ln = threadIdx.x & 31;
  int nl = ln & 15, mh = (ln >> 4) * 8;
#pragma unroll
  for (int mt = 0; mt < 4; ++mt)
#pragma unroll
    for (int n2 = 0; n2 < 4; ++n2) {
      int kg = wv * 64 + n2 * 16 + nl;
#pragma unroll
      for (int j = 0; j < 8; ++j)
        ldsA[afidx(mt, kg >> 5, j + mh, kg & 31)] = (__bf16)acc[mt][n2][j];
    }
}

// ---------------- front-end kernels (B=16 rows, negligible cost) ----------------
__global__ __launch_bounds__(256) void k_proj_in(const float* __restrict__ gnn,
                                                 const float* __restrict__ g,
                                                 const float* __restrict__ bt,
                                                 const float* __restrict__ W,
                                                 const float* __restrict__ bW,
                                                 float* __restrict__ h) {
  __shared__ float nrm[GNNF];
  __shared__ float red8[8];
  int b = blockIdx.x, tid = threadIdx.x;
  float x = gnn[b * GNNF + tid];
  float s  = block_sum256(x, red8);
  float ss = block_sum256(x * x, red8);
  float mean = s * (1.f / GNNF);
  float inv  = rsqrtf(ss * (1.f / GNNF) - mean * mean + 1e-5f);
  nrm[tid] = (x - mean) * inv * g[tid] + bt[tid];
  __syncthreads();
  for (int o = tid; o < HD; o += 256) {
    const float* wr = W + (size_t)o * GNNF;
    float a = bW[o];
    for (int i = 0; i < GNNF; ++i) a += nrm[i] * wr[i];
    h[b * HD + o] = gelu_f(a);
  }
}

__global__ __launch_bounds__(256) void k_res(float* __restrict__ h,
                                             const float* __restrict__ g,
                                             const float* __restrict__ bt,
                                             const float* __restrict__ W1,
                                             const float* __restrict__ b1,
                                             const float* __restrict__ W2,
                                             const float* __restrict__ b2) {
  __shared__ float y[HD];
  __shared__ float inner[INNERD];
  __shared__ float red8[8];
  int b = blockIdx.x, tid = threadIdx.x;
  float x0 = h[b * HD + tid], x1 = h[b * HD + 256 + tid];
  float s  = block_sum256(x0 + x1, red8);
  float ss = block_sum256(x0 * x0 + x1 * x1, red8);
  float mean = s * (1.f / HD);
  float inv  = rsqrtf(ss * (1.f / HD) - mean * mean + 1e-5f);
  y[tid]       = (x0 - mean) * inv * g[tid] + bt[tid];
  y[tid + 256] = (x1 - mean) * inv * g[tid + 256] + bt[tid + 256];
  __syncthreads();
  for (int o = tid; o < INNERD; o += 256) {
    const float* wr = W1 + (size_t)o * HD;
    float a = b1[o];
    for (int i = 0; i < HD; ++i) a += y[i] * wr[i];
    inner[o] = gelu_f(a);
  }
  __syncthreads();
  for (int o = tid; o < HD; o += 256) {
    const float* wr = W2 + (size_t)o * INNERD;
    float a = b2[o];
    for (int i = 0; i < INNERD; ++i) a += inner[i] * wr[i];
    h[b * HD + o] += a;
  }
}

// norm_pert LN -> ctx (2048) -> K/V for both attention layers
__global__ __launch_bounds__(256) void k_ctx_kv(const float* __restrict__ h,
                                                const float* __restrict__ g,
                                                const float* __restrict__ bt,
                                                const float* __restrict__ cW,
                                                const float* __restrict__ cb,
                                                const float* __restrict__ Wqkv0,
                                                const float* __restrict__ bqkv0,
                                                const float* __restrict__ Wqkv1,
                                                const float* __restrict__ bqkv1,
                                                float* __restrict__ kvbuf) {
  __shared__ float nr[HD];
  __shared__ float cx[CTXK * HD];
  __shared__ float red8[8];
  int b = blockIdx.x, tid = threadIdx.x;
  float x0 = h[b * HD + tid], x1 = h[b * HD + 256 + tid];
  float s  = block_sum256(x0 + x1, red8);
  float ss = block_sum256(x0 * x0 + x1 * x1, red8);
  float mean = s * (1.f / HD);
  float inv  = rsqrtf(ss * (1.f / HD) - mean * mean + 1e-5f);
  nr[tid]       = (x0 - mean) * inv * g[tid] + bt[tid];
  nr[tid + 256] = (x1 - mean) * inv * g[tid + 256] + bt[tid + 256];
  __syncthreads();
  for (int o = tid; o < CTXK * HD; o += 256) {
    const float* wr = cW + (size_t)o * HD;
    float a = cb[o];
    for (int i = 0; i < HD; ++i) a += nr[i] * wr[i];
    cx[o] = a;
  }
  __syncthreads();
  for (int l = 0; l < 2; ++l) {
    const float* Wq = l ? Wqkv1 : Wqkv0;
    const float* bq = l ? bqkv1 : bqkv0;
    for (int o = tid; o < CTXK * 2 * HD; o += 256) {
      int kk = o >> 10, rem = o & 1023, sel = rem >> 9, d = rem & 511;
      int wrow = HD + sel * HD + d;  // Wk rows 512..1023, Wv rows 1024..1535
      const float* wr = Wq + (size_t)wrow * HD;
      const float* cr = cx + kk * HD;
      float a = bq[wrow];
      for (int i = 0; i < HD; ++i) a += cr[i] * wr[i];
      kvbuf[((size_t)(l * BB + b) * CTXK + kk) * (2 * HD) + sel * HD + d] = a;
    }
  }
}

// ---------------- layer 0: batch-invariant Q = LN(gene_q) @ Wq^T + bq ----------------
__global__ __launch_bounds__(256) void k_l0_qgemm(const float* __restrict__ gene_q,
                                                  const float* __restrict__ g,
                                                  const float* __restrict__ bt,
                                                  const __bf16* __restrict__ wq,
                                                  const float* __restrict__ bq,
                                                  float* __restrict__ q0) {
  __shared__ __bf16 ldsA[4 * 16 * 512];
  int row0 = blockIdx.x * MT;
  ln_rows_to_ldsA(ldsA, gene_q, row0, NGENE, g, bt);
  __syncthreads();
  v8f z = {0.f, 0.f, 0.f, 0.f, 0.f, 0.f, 0.f, 0.f};
  v8f acc[4][4];
#pragma unroll
  for (int mt = 0; mt < 4; ++mt)
#pragma unroll
    for (int n2 = 0; n2 < 4; ++n2) acc[mt][n2] = z;
  gemm512(ldsA, wq, acc);
  int wv = threadIdx.x >> 5, ln = threadIdx.x & 31;
  int nl = ln & 15, mh = (ln >> 4) * 8;
#pragma unroll
  for (int mt = 0; mt < 4; ++mt)
#pragma unroll
    for (int n2 = 0; n2 < 4; ++n2) {
      int nc = wv * 64 + n2 * 16 + nl;
      float bv = bq[nc];
#pragma unroll
      for (int j = 0; j < 8; ++j) {
        int rg = row0 + mt * 16 + j + mh;
        if (rg < NGENE) q0[(size_t)rg * HD + nc] = acc[mt][n2][j] + bv;
      }
    }
}

// ---------------- layer 0: attention + O-projection + residual -> geneA[b] ----------------
__global__ __launch_bounds__(256) void k_attn_o(const float* __restrict__ q0,
                                                const float* __restrict__ kv,
                                                const float* __restrict__ gene_q,
                                                const __bf16* __restrict__ wo,
                                                const float* __restrict__ bo,
                                                float* __restrict__ gout) {
  __shared__ __bf16 ldsA[4 * 16 * 512];
  __shared__ float ldsKV[CTXK * 2 * HD];
  int row0 = blockIdx.x * MT;
  int b = blockIdx.y;
  // kick off K/V DMA into LDS, overlap with Q register loads
  tdm_issue_kv(ldsKV, kv + (size_t)b * CTXK * 2 * HD);
  int wv = threadIdx.x >> 5, ln = threadIdx.x & 31;
  int nl = ln & 15, mh = (ln >> 4) * 8;
  v8f acc[4][4];
#pragma unroll
  for (int mt = 0; mt < 4; ++mt)
#pragma unroll
    for (int n2 = 0; n2 < 4; ++n2) {
      int nc = wv * 64 + n2 * 16 + nl;
#pragma unroll
      for (int j = 0; j < 8; ++j) {
        int rg = row0 + mt * 16 + j + mh;
        int rc = rg < NGENE ? rg : NGENE - 1;
        acc[mt][n2][j] = q0[(size_t)rc * HD + nc];
      }
    }
  tdm_wait_kv();
  __syncthreads();
  attn4(acc, ldsKV);
  store_o_ldsA(ldsA, acc);
  __syncthreads();
  v8f z = {0.f, 0.f, 0.f, 0.f, 0.f, 0.f, 0.f, 0.f};
#pragma unroll
  for (int mt = 0; mt < 4; ++mt)
#pragma unroll
    for (int n2 = 0; n2 < 4; ++n2) acc[mt][n2] = z;
  gemm512(ldsA, wo, acc);
#pragma unroll
  for (int mt = 0; mt < 4; ++mt)
#pragma unroll
    for (int n2 = 0; n2 < 4; ++n2) {
      int nc = wv * 64 + n2 * 16 + nl;
      float bv = bo[nc];
#pragma unroll
      for (int j = 0; j < 8; ++j) {
        int rg = row0 + mt * 16 + j + mh;
        if (rg < NGENE) {
          size_t gi = ((size_t)b * NGENE + rg) * HD + nc;
          gout[gi] = acc[mt][n2][j] + bv + gene_q[(size_t)rg * HD + nc];
        }
      }
    }
}

// ---------------- layer 1: fully fused LN -> Q-WMMA -> attn -> O-WMMA -> residual (in place) ---------
__global__ __launch_bounds__(256) void k_l1_fused(float* __restrict__ gene,
                                                  const float* __restrict__ g,
                                                  const float* __restrict__ bt,
                                                  const __bf16* __restrict__ wq,
                                                  const float* __restrict__ bq,
                                                  const float* __restrict__ kv,
                                                  const __bf16* __restrict__ wo,
                                                  const float* __restrict__ bo) {
  __shared__ __bf16 ldsA[4 * 16 * 512];
  __shared__ float ldsKV[CTXK * 2 * HD];
  int row0 = blockIdx.x * MT;
  int b = blockIdx.y;
  const float* gbase = gene + (size_t)b * NGENE * HD;
  // kick off K/V DMA into LDS; it completes while we do LayerNorm
  tdm_issue_kv(ldsKV, kv + (size_t)b * CTXK * 2 * HD);
  ln_rows_to_ldsA(ldsA, gbase, row0, NGENE, g, bt);
  tdm_wait_kv();
  __syncthreads();
  v8f z = {0.f, 0.f, 0.f, 0.f, 0.f, 0.f, 0.f, 0.f};
  v8f acc[4][4];
#pragma unroll
  for (int mt = 0; mt < 4; ++mt)
#pragma unroll
    for (int n2 = 0; n2 < 4; ++n2) acc[mt][n2] = z;
  gemm512(ldsA, wq, acc);
  int wv = threadIdx.x >> 5, ln = threadIdx.x & 31;
  int nl = ln & 15, mh = (ln >> 4) * 8;
#pragma unroll
  for (int n2 = 0; n2 < 4; ++n2) {
    float bv = bq[wv * 64 + n2 * 16 + nl];
#pragma unroll
    for (int mt = 0; mt < 4; ++mt)
#pragma unroll
      for (int j = 0; j < 8; ++j) acc[mt][n2][j] += bv;
  }
  attn4(acc, ldsKV);
  __syncthreads();             // all waves done reading Q-GEMM LDS tile
  store_o_ldsA(ldsA, acc);
  __syncthreads();
#pragma unroll
  for (int mt = 0; mt < 4; ++mt)
#pragma unroll
    for (int n2 = 0; n2 < 4; ++n2) acc[mt][n2] = z;
  gemm512(ldsA, wo, acc);
#pragma unroll
  for (int mt = 0; mt < 4; ++mt)
#pragma unroll
    for (int n2 = 0; n2 < 4; ++n2) {
      int nc = wv * 64 + n2 * 16 + nl;
      float bv = bo[nc];
#pragma unroll
      for (int j = 0; j < 8; ++j) {
        int rg = row0 + mt * 16 + j + mh;
        if (rg < NGENE) {
          size_t gi = ((size_t)b * NGENE + rg) * HD + nc;
          gene[gi] = acc[mt][n2][j] + bv + gbase[(size_t)rg * HD + nc];
        }
      }
    }
}

// ---------------- out_norm + classifier (512 -> 3), one wave per gene row ----------------
__global__ __launch_bounds__(256) void k_out(const float* __restrict__ gene,
                                             const float* __restrict__ g,
                                             const float* __restrict__ bt,
                                             const float* __restrict__ cW,
                                             const float* __restrict__ cb,
                                             float* __restrict__ out) {
  int wv = threadIdx.x >> 5, ln = threadIdx.x & 31;
  int gw = blockIdx.x * 8 + wv;
  if (gw >= BB * NGENE) return;
  int b = gw / NGENE, r = gw - b * NGENE;
  const float* src = gene + (size_t)gw * HD;
  int c0 = ln * 16;
  float vals[16];
  float s = 0.f, ss = 0.f;
#pragma unroll
  for (int j = 0; j < 16; ++j) { float x = src[c0 + j]; vals[j] = x; s += x; ss += x * x; }
#pragma unroll
  for (int m = 1; m < 32; m <<= 1) { s += __shfl_xor(s, m); ss += __shfl_xor(ss, m); }
  float mean = s * (1.f / HD);
  float inv  = rsqrtf(ss * (1.f / HD) - mean * mean + 1e-5f);
  float l0 = 0.f, l1 = 0.f, l2 = 0.f;
#pragma unroll
  for (int j = 0; j < 16; ++j) {
    float y = (vals[j] - mean) * inv * g[c0 + j] + bt[c0 + j];
    l0 += y * cW[0 * HD + c0 + j];
    l1 += y * cW[1 * HD + c0 + j];
    l2 += y * cW[2 * HD + c0 + j];
  }
#pragma unroll
  for (int m = 1; m < 32; m <<= 1) {
    l0 += __shfl_xor(l0, m); l1 += __shfl_xor(l1, m); l2 += __shfl_xor(l2, m);
  }
  if (ln == 0) {
    out[((size_t)b * NCLS + 0) * NGENE + r] = l0 + cb[0];
    out[((size_t)b * NCLS + 1) * NGENE + r] = l1 + cb[1];
    out[((size_t)b * NCLS + 2) * NGENE + r] = l2 + cb[2];
  }
}

// ---------------- host-side orchestration ----------------
extern "C" void kernel_launch(void* const* d_in, const int* in_sizes, int n_in,
                              void* d_out, int out_size, void* d_ws, size_t ws_size,
                              hipStream_t stream) {
  (void)in_sizes; (void)n_in; (void)out_size; (void)ws_size;
  auto F = [&](int i) { return (const float*)d_in[i]; };

  // input order per setup_inputs() dict traversal
  const float* gnn   = F(0);
  const float* pig   = F(1);  const float* pib = F(2);
  const float* piW   = F(3);  const float* pibW = F(4);
  // res blocks: 5 + 6*r : g,b,W1,b1,W2,b2
  const float* npg   = F(29); const float* npb = F(30);
  const float* ctxW  = F(31); const float* ctxb = F(32);
  const float* geneq = F(33);
  const float* ag0 = F(34), *ab0 = F(35), *aWqkv0 = F(36), *abqkv0 = F(37), *aWo0 = F(38), *abo0 = F(39);
  const float* ag1 = F(40), *ab1 = F(41), *aWqkv1 = F(42), *abqkv1 = F(43), *aWo1 = F(44), *abo1 = F(45);
  const float* ong = F(46), *onb = F(47), *clsW = F(48), *clsb = F(49);

  // workspace layout (floats, then bf16 packed weights; offsets keep 32B alignment)
  float* wh    = (float*)d_ws;                           // [16,512]
  float* kvbuf = wh + (size_t)BB * HD;                   // [2,16,4,2,512]
  float* q0    = kvbuf + (size_t)2 * BB * CTXK * 2 * HD; // [6640,512]
  float* geneA = q0 + (size_t)NGENE * HD;                // [16,6640,512] (updated in place by layer 1)
  __bf16* wpk  = (__bf16*)(geneA + (size_t)BB * NGENE * HD); // 4 x [512*512] bf16 fragments

  const size_t WSZ = (size_t)HD * HD;  // 262144 elems per packed matrix

  // pack Wq/Wo of both layers into WMMA B-fragment layout (Wq = first 512 rows of Wqkv)
  k_pack_w<<<1024, 256, 0, stream>>>(aWqkv0, wpk + 0 * WSZ);
  k_pack_w<<<1024, 256, 0, stream>>>(aWo0,   wpk + 1 * WSZ);
  k_pack_w<<<1024, 256, 0, stream>>>(aWqkv1, wpk + 2 * WSZ);
  k_pack_w<<<1024, 256, 0, stream>>>(aWo1,   wpk + 3 * WSZ);

  // tiny front-end (B=16 rows)
  k_proj_in<<<BB, 256, 0, stream>>>(gnn, pig, pib, piW, pibW, wh);
  for (int r = 0; r < 4; ++r) {
    int base = 5 + 6 * r;
    k_res<<<BB, 256, 0, stream>>>(wh, F(base), F(base + 1), F(base + 2),
                                  F(base + 3), F(base + 4), F(base + 5));
  }
  k_ctx_kv<<<BB, 256, 0, stream>>>(wh, npg, npb, ctxW, ctxb,
                                   aWqkv0, abqkv0, aWqkv1, abqkv1, kvbuf);

  // layer 0: batch-invariant Q once, then per-batch attention + O-proj
  k_l0_qgemm<<<TILES, 256, 0, stream>>>(geneq, ag0, ab0, wpk + 0 * WSZ, abqkv0, q0);
  k_attn_o<<<dim3(TILES, BB), 256, 0, stream>>>(q0, kvbuf, geneq,
                                                wpk + 1 * WSZ, abo0, geneA);

  // layer 1: fully fused, in-place gene update
  const float* kv1 = kvbuf + (size_t)BB * CTXK * 2 * HD;
  k_l1_fused<<<dim3(TILES, BB), 256, 0, stream>>>(geneA, ag1, ab1, wpk + 2 * WSZ, abqkv1,
                                                  kv1, wpk + 3 * WSZ, abo1);

  // out_norm + classifier, output [B, NC, NG]
  k_out<<<(BB * NGENE) / 8, 256, 0, stream>>>(geneA, ong, onb, clsW, clsb, (float*)d_out);
}
